// LongContextGenomicModel_27934467293645
// MI455X (gfx1250) — compile-verified, matrix operand order checked by reading, not run on hardware
//
#include <hip/hip_runtime.h>
#include <hip/hip_bf16.h>
#include <math.h>

// ---------------- model constants ----------------
#define L_SEQ   16384
#define DM_     256
#define DI_     512
#define NST     16
#define RLOW    32
#define NLAYERS 6
#define KCONV   4

// scan chunking: 128 chunks of 128 steps
#define NCHUNK  128
#define TCHUNK  128
#define SSTATES (DI_ * NST)   // 8192

typedef __attribute__((ext_vector_type(16))) __bf16 v16bf;
typedef __attribute__((ext_vector_type(8)))  float  v8f;

union FragBf {
    v16bf v;
    uint4 q[2];
};

// ---------------- small helpers ----------------
__device__ __forceinline__ float softplus_f(float x) {
    return (x > 20.f) ? x : log1pf(__expf(x));
}
__device__ __forceinline__ float silu_f(float x) {
    return x / (1.f + __expf(-x));
}

// CDNA5 async memory->LDS copy (16 bytes per lane), tracked by ASYNCcnt.
// The low 32 bits of a generic __shared__ pointer are the LDS byte offset.
__device__ __forceinline__ void async_copy16(void* lds_dst, const void* gsrc) {
    unsigned int loff = (unsigned int)(unsigned long long)lds_dst;
    unsigned long long g = (unsigned long long)gsrc;
    asm volatile("global_load_async_to_lds_b128 %0, %1, off"
                 :: "v"(loff), "v"(g) : "memory");
}
__device__ __forceinline__ void async_wait0() {
    asm volatile("s_wait_asynccnt 0x0" ::: "memory");
}

// ---------------- input embed: h = concat(x,meth)^T @ Wi + b ----------------
__global__ void embed_kernel(const float* __restrict__ x,
                             const float* __restrict__ meth,
                             const float* __restrict__ wi,   // (8, DM)
                             const float* __restrict__ bi,   // (DM)
                             float* __restrict__ h) {        // (L, DM)
    int idx = blockIdx.x * blockDim.x + threadIdx.x;
    if (idx >= L_SEQ * DM_) return;
    int l = idx >> 8;        // /256
    int d = idx & 255;
    float acc = bi[d];
#pragma unroll
    for (int c = 0; c < 5; ++c) acc += x[c * L_SEQ + l] * wi[c * DM_ + d];
#pragma unroll
    for (int c = 0; c < 3; ++c) acc += meth[c * L_SEQ + l] * wi[(5 + c) * DM_ + d];
    h[idx] = acc;
}

// ---------------- fp32 (K,N) -> bf16 transposed (N,K) ----------------
__global__ void wconv_kernel(const float* __restrict__ W, __hip_bfloat16* __restrict__ Wt,
                             int K, int N) {
    int idx = blockIdx.x * blockDim.x + threadIdx.x;
    if (idx >= K * N) return;
    int k = idx / N;
    int n = idx - k * N;
    Wt[n * K + k] = __float2bfloat16(W[idx]);
}

// ---------------- LayerNorm (per row of DM=256) -> bf16 ----------------
// block = 256 threads = 8 waves, one row per wave
__global__ void ln_kernel(const float* __restrict__ h,
                          const float* __restrict__ g, const float* __restrict__ b,
                          __hip_bfloat16* __restrict__ xn) {
    int wave = threadIdx.x >> 5;
    int lane = threadIdx.x & 31;
    int row = blockIdx.x * 8 + wave;
    const float* hr = h + (size_t)row * DM_;
    float v[8], s = 0.f, ss = 0.f;
#pragma unroll
    for (int j = 0; j < 8; ++j) {
        v[j] = hr[lane + j * 32];
        s += v[j]; ss += v[j] * v[j];
    }
#pragma unroll
    for (int m = 1; m < 32; m <<= 1) {
        s  += __shfl_xor(s,  m);
        ss += __shfl_xor(ss, m);
    }
    float mu = s * (1.f / DM_);
    float var = ss * (1.f / DM_) - mu * mu;
    float rstd = rsqrtf(var + 1e-5f);
    __hip_bfloat16* xr = xn + (size_t)row * DM_;
#pragma unroll
    for (int j = 0; j < 8; ++j) {
        int c = lane + j * 32;
        xr[c] = __float2bfloat16((v[j] - mu) * rstd * g[c] + b[c]);
    }
}

// ---------------- WMMA bf16 GEMM: C(M,N) = A(M,K) @ Bt(N,K)^T ----------------
// block = 128 threads (4 waves); wave w handles tileN = blockIdx.y*4 + w
// modes: 0 = store f32; 1 = store bf16; 2 = +bias, softplus -> bf16;
//        3 = += residual (in place, f32)
__global__ void wmma_gemm_kernel(const __hip_bfloat16* __restrict__ A,
                                 const __hip_bfloat16* __restrict__ Bt,
                                 int K, int ldc, int mode,
                                 const float* __restrict__ bias,
                                 const float* __restrict__ resid,
                                 float* __restrict__ outF,
                                 __hip_bfloat16* __restrict__ outB) {
    int wave = threadIdx.x >> 5;
    int lane = threadIdx.x & 31;
    int tileM = blockIdx.x;
    int tileN = blockIdx.y * 4 + wave;

    int r16 = lane & 15;
    int kh  = lane >> 4;           // K-half select for A/B frags, M-half for C/D

    const __hip_bfloat16* pA = A  + (size_t)(tileM * 16 + r16) * K;
    const __hip_bfloat16* pB = Bt + (size_t)(tileN * 16 + r16) * K;

    v8f acc = {0.f, 0.f, 0.f, 0.f, 0.f, 0.f, 0.f, 0.f};

    for (int k0 = 0; k0 < K; k0 += 32) {
        FragBf fa, fb;
        fa.q[0] = *reinterpret_cast<const uint4*>(pA + k0 + kh * 8);
        fa.q[1] = *reinterpret_cast<const uint4*>(pA + k0 + 16 + kh * 8);
        fb.q[0] = *reinterpret_cast<const uint4*>(pB + k0 + kh * 8);
        fb.q[1] = *reinterpret_cast<const uint4*>(pB + k0 + 16 + kh * 8);
        if (k0 + 32 < K) {
            __builtin_prefetch(pA + k0 + 64, 0, 3);   // -> global_prefetch_b8 (near)
            __builtin_prefetch(pB + k0 + 64, 0, 3);
        }
        acc = __builtin_amdgcn_wmma_f32_16x16x32_bf16(
                  false, fa.v, false, fb.v, (short)0, acc, false, false);
    }

    int col = tileN * 16 + (lane & 15);
#pragma unroll
    for (int r = 0; r < 8; ++r) {
        int m = tileM * 16 + r + 8 * kh;
        size_t idx = (size_t)m * ldc + col;
        float v = acc[r];
        if (mode == 0) {
            outF[idx] = v;
        } else if (mode == 1) {
            outB[idx] = __float2bfloat16(v);
        } else if (mode == 2) {
            outB[idx] = __float2bfloat16(softplus_f(v + bias[col]));
        } else { // mode 3: residual add in place
            outF[idx] = resid[idx] + v;
        }
    }
}

// ---------------- causal depthwise conv (K=4) + SiLU -> u (bf16) ----------------
// xz: (L, 2*DI) bf16, xb = cols [0,DI)
__global__ void conv_silu_kernel(const __hip_bfloat16* __restrict__ xz,
                                 const float* __restrict__ cw,   // (DI, 4)
                                 const float* __restrict__ cb,   // (DI)
                                 __hip_bfloat16* __restrict__ u) {
    int idx = blockIdx.x * blockDim.x + threadIdx.x;
    if (idx >= L_SEQ * DI_) return;
    int l = idx >> 9;        // /512
    int d = idx & 511;
    float acc = cb[d];
#pragma unroll
    for (int k = 0; k < KCONV; ++k) {
        int ll = l - (KCONV - 1) + k;
        if (ll >= 0)
            acc += __bfloat162float(xz[(size_t)ll * (2 * DI_) + d]) * cw[d * KCONV + k];
    }
    u[idx] = __float2bfloat16(silu_f(acc));
}

// ---------------- dt_low slice (first RLOW cols of xdbl) -> bf16 ----------------
__global__ void dtl_kernel(const float* __restrict__ xdbl, __hip_bfloat16* __restrict__ dtl) {
    int idx = blockIdx.x * blockDim.x + threadIdx.x;
    if (idx >= L_SEQ * RLOW) return;
    int l = idx >> 5;
    int r = idx & 31;
    dtl[idx] = __float2bfloat16(xdbl[(size_t)l * (RLOW + 2 * NST) + r]);
}

// ---------------- scan pass A: per-chunk decay product + local state ----------------
// grid = (NCHUNK, DI/16); block = 256: tid = dl*16 + n
// dt/u tiles staged in LDS as raw bf16 via async copies (16B/lane); B tile f32.
__global__ void scan_partial_kernel(const __hip_bfloat16* __restrict__ dt,   // (L, DI)
                                    const __hip_bfloat16* __restrict__ u,    // (L, DI)
                                    const float* __restrict__ xdbl,          // (L, 64)
                                    const float* __restrict__ A_log,         // (DI, NST)
                                    float* __restrict__ aprodC,              // (NCHUNK, SSTATES)
                                    float* __restrict__ hlocC) {             // (NCHUNK, SSTATES)
    __shared__ __hip_bfloat16 s_dt16[TCHUNK * 16];
    __shared__ __hip_bfloat16 s_u16 [TCHUNK * 16];
    __shared__ float          s_B   [TCHUNK * NST];

    int tid = threadIdx.x;
    int n  = tid & 15;
    int dl = tid >> 4;
    int c  = blockIdx.x;
    int dbase = blockIdx.y * 16;
    int d = dbase + dl;
    int l0 = c * TCHUNK;

    // dt/u: 128 rows x 32B  -> 256 x 16B async copies each
    {
        int t = tid >> 1, half = tid & 1;
        async_copy16(&s_dt16[t * 16 + half * 8],
                     dt + (size_t)(l0 + t) * DI_ + dbase + half * 8);
        async_copy16(&s_u16[t * 16 + half * 8],
                     u + (size_t)(l0 + t) * DI_ + dbase + half * 8);
    }
    // B: 128 rows x 64B -> 512 x 16B async copies (2 per thread)
#pragma unroll
    for (int j = 0; j < 2; ++j) {
        int e = tid + j * 256;
        int t = e >> 2, q = e & 3;
        async_copy16(&s_B[t * NST + q * 4],
                     xdbl + (size_t)(l0 + t) * (RLOW + 2 * NST) + RLOW + q * 4);
    }
    async_wait0();
    __syncthreads();

    float Aval = -__expf(A_log[d * NST + n]);
    float hv = 0.f, ap = 1.f;
#pragma unroll 4
    for (int t = 0; t < TCHUNK; ++t) {
        float dtv = __bfloat162float(s_dt16[t * 16 + dl]);
        float uv  = __bfloat162float(s_u16 [t * 16 + dl]);
        float Bv  = s_B[t * NST + n];
        float a = __expf(dtv * Aval);
        hv = a * hv + dtv * uv * Bv;
        ap *= a;
    }
    size_t s = (size_t)c * SSTATES + d * NST + n;
    aprodC[s] = ap;
    hlocC[s] = hv;
}

// ---------------- scan pass B: prefix over chunks (8192-way parallel) ----------------
__global__ void scan_prefix_kernel(const float* __restrict__ aprodC,
                                   const float* __restrict__ hlocC,
                                   float* __restrict__ hinitC) {
    int s = blockIdx.x * blockDim.x + threadIdx.x;
    if (s >= SSTATES) return;
    float hv = 0.f;
    for (int c = 0; c < NCHUNK; ++c) {
        size_t i = (size_t)c * SSTATES + s;
        hinitC[i] = hv;
        hv = aprodC[i] * hv + hlocC[i];
    }
}

// ---------------- scan pass C: replay + reduce over N + SiLU gate -> yg bf16 ----------
__global__ void scan_final_kernel(const __hip_bfloat16* __restrict__ dt,
                                  const __hip_bfloat16* __restrict__ u,
                                  const float* __restrict__ xdbl,
                                  const __hip_bfloat16* __restrict__ xz,    // z = cols [DI,2DI)
                                  const float* __restrict__ A_log,
                                  const float* __restrict__ Dp,             // (DI)
                                  const float* __restrict__ hinitC,
                                  __hip_bfloat16* __restrict__ yg) {        // (L, DI)
    __shared__ __hip_bfloat16 s_dt16[TCHUNK * 16];
    __shared__ __hip_bfloat16 s_u16 [TCHUNK * 16];
    __shared__ float          s_B   [TCHUNK * NST];
    __shared__ float          s_C   [TCHUNK * NST];

    int tid = threadIdx.x;
    int n  = tid & 15;
    int dl = tid >> 4;
    int c  = blockIdx.x;
    int dbase = blockIdx.y * 16;
    int d = dbase + dl;
    int l0 = c * TCHUNK;

    {
        int t = tid >> 1, half = tid & 1;
        async_copy16(&s_dt16[t * 16 + half * 8],
                     dt + (size_t)(l0 + t) * DI_ + dbase + half * 8);
        async_copy16(&s_u16[t * 16 + half * 8],
                     u + (size_t)(l0 + t) * DI_ + dbase + half * 8);
    }
#pragma unroll
    for (int j = 0; j < 2; ++j) {
        int e = tid + j * 256;
        int t = e >> 2, q = e & 3;
        async_copy16(&s_B[t * NST + q * 4],
                     xdbl + (size_t)(l0 + t) * (RLOW + 2 * NST) + RLOW + q * 4);
        async_copy16(&s_C[t * NST + q * 4],
                     xdbl + (size_t)(l0 + t) * (RLOW + 2 * NST) + RLOW + NST + q * 4);
    }
    async_wait0();
    __syncthreads();

    float Aval = -__expf(A_log[d * NST + n]);
    float dpv  = Dp[d];
    float hv = hinitC[(size_t)c * SSTATES + d * NST + n];

    for (int t = 0; t < TCHUNK; ++t) {
        float dtv = __bfloat162float(s_dt16[t * 16 + dl]);
        float uv  = __bfloat162float(s_u16 [t * 16 + dl]);
        float a = __expf(dtv * Aval);
        hv = a * hv + dtv * uv * s_B[t * NST + n];
        float p = hv * s_C[t * NST + n];
        // reduce over n (16 lanes; xor of bits 0..3 stays inside the group)
        p += __shfl_xor(p, 1);
        p += __shfl_xor(p, 2);
        p += __shfl_xor(p, 4);
        p += __shfl_xor(p, 8);
        if (n == 0) {
            float y = p + dpv * uv;
            float zf = __bfloat162float(xz[(size_t)(l0 + t) * (2 * DI_) + DI_ + d]);
            yg[(size_t)(l0 + t) * DI_ + d] = __float2bfloat16(y * silu_f(zf));
        }
    }
}

// ---------------- final LN + mean-pool (accumulate into embed[DM]) ----------------
__global__ void zero_embed_kernel(float* __restrict__ embed) {
    embed[threadIdx.x] = 0.f;
}

__global__ void ln_pool_kernel(const float* __restrict__ h,
                               const float* __restrict__ g, const float* __restrict__ b,
                               float* __restrict__ embed) {
    __shared__ float s_acc[DM_];
    s_acc[threadIdx.x] = 0.f;
    __syncthreads();

    int wave = threadIdx.x >> 5;
    int lane = threadIdx.x & 31;
    int row = blockIdx.x * 8 + wave;
    const float* hr = h + (size_t)row * DM_;
    float v[8], s = 0.f, ss = 0.f;
#pragma unroll
    for (int j = 0; j < 8; ++j) {
        v[j] = hr[lane + j * 32];
        s += v[j]; ss += v[j] * v[j];
    }
#pragma unroll
    for (int m = 1; m < 32; m <<= 1) {
        s  += __shfl_xor(s,  m);
        ss += __shfl_xor(ss, m);
    }
    float mu = s * (1.f / DM_);
    float rstd = rsqrtf(ss * (1.f / DM_) - mu * mu + 1e-5f);
#pragma unroll
    for (int j = 0; j < 8; ++j) {
        int cidx = lane + j * 32;
        atomicAdd(&s_acc[cidx], (v[j] - mu) * rstd * g[cidx] + b[cidx]);
    }
    __syncthreads();
    atomicAdd(&embed[threadIdx.x], s_acc[threadIdx.x]);
}

// ---------------- head: gelu(embed/L @ W1 + b1) @ W2 + b2 ----------------
__global__ void head_kernel(const float* __restrict__ embed,
                            const float* __restrict__ w1, const float* __restrict__ b1,
                            const float* __restrict__ w2, const float* __restrict__ b2,
                            float* __restrict__ out) {
    __shared__ float s_part[128];
    int j = threadIdx.x;           // 0..127
    float acc = b1[j];
    const float scale = 1.f / (float)L_SEQ;
    for (int d = 0; d < DM_; ++d)
        acc += embed[d] * scale * w1[d * 128 + j];
    float hid = 0.5f * acc * (1.f + erff(acc * 0.70710678118f));
    s_part[j] = hid * w2[j];
    __syncthreads();
    for (int str = 64; str > 0; str >>= 1) {
        if (j < str) s_part[j] += s_part[j + str];
        __syncthreads();
    }
    if (j == 0) out[0] = s_part[0] + b2[0];
}

// =======================================================================
extern "C" void kernel_launch(void* const* d_in, const int* in_sizes, int n_in,
                              void* d_out, int out_size, void* d_ws, size_t ws_size,
                              hipStream_t stream) {
    const float* x        = (const float*)d_in[0];
    const float* meth     = (const float*)d_in[1];
    const float* inp_w    = (const float*)d_in[2];
    const float* inp_b    = (const float*)d_in[3];
    const float* norm_g   = (const float*)d_in[4];
    const float* norm_b   = (const float*)d_in[5];
    const float* in_proj  = (const float*)d_in[6];
    const float* conv_w   = (const float*)d_in[7];
    const float* conv_b   = (const float*)d_in[8];
    const float* xproj_w  = (const float*)d_in[9];
    const float* dtproj_w = (const float*)d_in[10];
    const float* dtproj_b = (const float*)d_in[11];
    const float* A_log    = (const float*)d_in[12];
    const float* Dp       = (const float*)d_in[13];
    const float* outproj  = (const float*)d_in[14];
    const float* fn_g     = (const float*)d_in[15];
    const float* fn_b     = (const float*)d_in[16];
    const float* head_w1  = (const float*)d_in[17];
    const float* head_b1  = (const float*)d_in[18];
    const float* head_w2  = (const float*)d_in[19];
    const float* head_b2  = (const float*)d_in[20];
    float* out = (float*)d_out;

    // workspace carve-up (deterministic, 256B aligned)
    char* ws = (char*)d_ws;
    size_t o = 0;
    auto alloc = [&](size_t bytes) { size_t r = o; o = (o + bytes + 255) & ~(size_t)255; return r; };
    const size_t L = L_SEQ;
    float*           h     = (float*)          (ws + alloc(L * DM_ * 4));
    __hip_bfloat16*  xn    = (__hip_bfloat16*) (ws + alloc(L * DM_ * 2));
    __hip_bfloat16*  xz    = (__hip_bfloat16*) (ws + alloc(L * 2 * DI_ * 2));
    __hip_bfloat16*  u     = (__hip_bfloat16*) (ws + alloc(L * DI_ * 2));
    float*           xdbl  = (float*)          (ws + alloc(L * (RLOW + 2 * NST) * 4));
    __hip_bfloat16*  dtl   = (__hip_bfloat16*) (ws + alloc(L * RLOW * 2));
    __hip_bfloat16*  dtb   = (__hip_bfloat16*) (ws + alloc(L * DI_ * 2));
    __hip_bfloat16*  yg    = (__hip_bfloat16*) (ws + alloc(L * DI_ * 2));
    float*           aprod = (float*)          (ws + alloc((size_t)NCHUNK * SSTATES * 4));
    float*           hloc  = (float*)          (ws + alloc((size_t)NCHUNK * SSTATES * 4));
    float*           hinit = (float*)          (ws + alloc((size_t)NCHUNK * SSTATES * 4));
    __hip_bfloat16*  wi_t  = (__hip_bfloat16*) (ws + alloc((size_t)DM_ * 2 * DI_ * 2));
    __hip_bfloat16*  xp_t  = (__hip_bfloat16*) (ws + alloc((size_t)DI_ * (RLOW + 2 * NST) * 2));
    __hip_bfloat16*  dtp_t = (__hip_bfloat16*) (ws + alloc((size_t)RLOW * DI_ * 2));
    __hip_bfloat16*  op_t  = (__hip_bfloat16*) (ws + alloc((size_t)DI_ * DM_ * 2));
    float*           embed = (float*)          (ws + alloc(DM_ * 4));
    (void)ws_size; (void)in_sizes; (void)n_in; (void)out_size;

    // 1) input embed
    embed_kernel<<<(L * DM_ + 255) / 256, 256, 0, stream>>>(x, meth, inp_w, inp_b, h);

    for (int i = 0; i < NLAYERS; ++i) {
        // weight converts (fp32 row-major -> bf16 transposed)
        wconv_kernel<<<(DM_ * 2 * DI_ + 255) / 256, 256, 0, stream>>>(
            in_proj + (size_t)i * DM_ * 2 * DI_, wi_t, DM_, 2 * DI_);
        wconv_kernel<<<(DI_ * (RLOW + 2 * NST) + 255) / 256, 256, 0, stream>>>(
            xproj_w + (size_t)i * DI_ * (RLOW + 2 * NST), xp_t, DI_, RLOW + 2 * NST);
        wconv_kernel<<<(RLOW * DI_ + 255) / 256, 256, 0, stream>>>(
            dtproj_w + (size_t)i * RLOW * DI_, dtp_t, RLOW, DI_);
        wconv_kernel<<<(DI_ * DM_ + 255) / 256, 256, 0, stream>>>(
            outproj + (size_t)i * DI_ * DM_, op_t, DI_, DM_);

        // LayerNorm -> bf16
        ln_kernel<<<L / 8, 256, 0, stream>>>(h, norm_g + i * DM_, norm_b + i * DM_, xn);

        // xz = xn @ in_proj  (L x 256) @ (256 x 1024) -> bf16
        wmma_gemm_kernel<<<dim3(L / 16, (2 * DI_) / 64), 128, 0, stream>>>(
            xn, wi_t, DM_, 2 * DI_, /*mode=*/1, nullptr, nullptr, nullptr, xz);

        // causal depthwise conv + SiLU -> u (bf16)
        conv_silu_kernel<<<(L * DI_ + 255) / 256, 256, 0, stream>>>(
            xz, conv_w + (size_t)i * DI_ * KCONV, conv_b + i * DI_, u);

        // xdbl = u @ xproj  (L x 512) @ (512 x 64) -> fp32
        wmma_gemm_kernel<<<dim3(L / 16, (RLOW + 2 * NST) / 64), 128, 0, stream>>>(
            u, xp_t, DI_, RLOW + 2 * NST, /*mode=*/0, nullptr, nullptr, xdbl, nullptr);

        // dt_low slice -> bf16
        dtl_kernel<<<(L * RLOW + 255) / 256, 256, 0, stream>>>(xdbl, dtl);

        // dt = softplus(dt_low @ dtproj + b)  (L x 32) @ (32 x 512) -> bf16
        wmma_gemm_kernel<<<dim3(L / 16, DI_ / 64), 128, 0, stream>>>(
            dtl, dtp_t, RLOW, DI_, /*mode=*/2, dtproj_b + i * DI_, nullptr, nullptr, dtb);

        // chunked selective scan (3 passes)
        scan_partial_kernel<<<dim3(NCHUNK, DI_ / 16), 256, 0, stream>>>(
            dtb, u, xdbl, A_log + (size_t)i * DI_ * NST, aprod, hloc);
        scan_prefix_kernel<<<SSTATES / 256, 256, 0, stream>>>(aprod, hloc, hinit);
        scan_final_kernel<<<dim3(NCHUNK, DI_ / 16), 256, 0, stream>>>(
            dtb, u, xdbl, xz, A_log + (size_t)i * DI_ * NST, Dp + i * DI_, hinit, yg);

        // h = yg @ outproj + h  (L x 512) @ (512 x 256), residual in place
        wmma_gemm_kernel<<<dim3(L / 16, DM_ / 64), 128, 0, stream>>>(
            yg, op_t, DI_, DM_, /*mode=*/3, nullptr, h, h, nullptr);
    }

    // final LN + mean pool + head
    zero_embed_kernel<<<1, DM_, 0, stream>>>(embed);
    ln_pool_kernel<<<L / 8, 256, 0, stream>>>(h, fn_g, fn_b, embed);
    head_kernel<<<1, 128, 0, stream>>>(embed, head_w1, head_b1, head_w2, head_b2, out);
}
